// HunyuanSplashAttnProcessor_90606630076907
// MI455X (gfx1250) — matrix operand order repaired and checked by reading, compiled, not run on hardware
//
#include <hip/hip_runtime.h>

// ---------------------------------------------------------------------------
// HunyuanVideo-style joint attention for MI455X (gfx1250, wave32, WMMA).
//   1) gemm_splitbf16: Q/K (fp32 out), V (bf16, transposed per head [H*128][S]).
//      fp32 operands split into bf16 hi/lo -> 3x v_wmma_f32_16x16x32_bf16.
//      LDS-staged tiles; all fragment reads are 2x ds_load_b128.
//   2) normrope: per-head RMSNorm + interleaved RoPE (first 2048 rows).
//   3) flash: 2 waves/block (2 query tiles, same head) sharing double-buffered
//      K/V chunks streamed with global_load_async_to_lds_b128 (ASYNCcnt),
//      online softmax (base 2), K/V/P WMMA fragments via ds_load_b128.
//   4) gemm_splitbf16: img = O[:2048]@wo + bo -> d_out.
// ---------------------------------------------------------------------------

#define DEVFN static __device__ __forceinline__

#define S_IMG   2048
#define S_TXT   256
#define S_ALL   2304
#define NHEADS  24
#define HD      128
#define DMODEL  3072

typedef __attribute__((ext_vector_type(16))) __bf16 v16bf;
typedef __attribute__((ext_vector_type(8)))  float  v8f;

union Frag16 { v16bf v; unsigned short u[16]; uint4 q[2]; };

DEVFN unsigned short f2bf(float f) {          // fp32 -> bf16 (RNE), bit-level
  unsigned int x = __float_as_uint(f);
  x += 0x7FFFu + ((x >> 16) & 1u);
  return (unsigned short)(x >> 16);
}
DEVFN float bf2f(unsigned short h) {
  return __uint_as_float(((unsigned int)h) << 16);
}

DEVFN v8f wmma_bf16(v16bf a, v16bf b, v8f c) {
  return __builtin_amdgcn_wmma_f32_16x16x32_bf16(
      /*neg_a=*/false, a, /*neg_b=*/false, b,
      /*c_mod=*/(short)0, c, /*reuse_a=*/false, /*reuse_b=*/false);
}

// A-frag: lane half g reads k in [8g, 8g+8) then [8g+16, 8g+24)  (2x b128)
DEVFN void load_afrag(Frag16& f, const unsigned short* base, int g) {
  f.q[0] = *reinterpret_cast<const uint4*>(base + 8 * g);
  f.q[1] = *reinterpret_cast<const uint4*>(base + 8 * g + 16);
}
// B-frag: lane half g reads k in [8g, 8g+8) then [8g+16, 8g+24)  (2x b128)
DEVFN void load_bfrag(Frag16& f, const unsigned short* base, int g) {
  f.q[0] = *reinterpret_cast<const uint4*>(base + 8 * g);
  f.q[1] = *reinterpret_cast<const uint4*>(base + 8 * g + 16);
}

// Async DMA: copy 16B per lane from global to LDS (tracked by ASYNCcnt).
DEVFN void async_copy_b128(void* lds, const void* gptr) {
  unsigned lds_off = (unsigned)(unsigned long long)lds;   // low 32b = LDS addr
  asm volatile("global_load_async_to_lds_b128 %0, %1, off"
               :: "v"(lds_off), "v"(gptr) : "memory");
}
DEVFN void wait_async0() {
  asm volatile("s_wait_asynccnt 0x0" ::: "memory");
}

// ---------------------------------------------------------------------------
// Split-bf16 GEMM:  out[M,N] = concat(A0,A1)[M,K] @ B[K,N] + bias
// Block = 256 threads (8 waves as 4x2), block tile 64x128, wave tile 16x64.
// sA: [row][k] (pad 40), sB: [col][k] transposed (pad 40) -> b128 frag loads.
// ---------------------------------------------------------------------------
__global__ __launch_bounds__(256)
void gemm_splitbf16_kernel(const float* __restrict__ A0,
                           const float* __restrict__ A1,
                           int rowsA0,
                           const float* __restrict__ B,
                           const float* __restrict__ bias,
                           float*          __restrict__ outF,   // may be null
                           unsigned short* __restrict__ outB,   // may be null, [M][3072]
                           unsigned short* __restrict__ outBT,  // may be null, [3072][S_ALL]
                           int M, int N, int K)
{
  __shared__ alignas(16) unsigned short sAhi[64][40];
  __shared__ alignas(16) unsigned short sAlo[64][40];
  __shared__ alignas(16) unsigned short sBhi[128][40];
  __shared__ alignas(16) unsigned short sBlo[128][40];

  const int tid   = threadIdx.x;
  const int lane  = tid & 31;
  const int wave  = tid >> 5;          // 0..7
  const int wm    = wave >> 1;         // 0..3  (M direction)
  const int wn    = wave & 1;          // 0..1  (N direction)
  const int mlane = lane & 15;
  const int g     = lane >> 4;

  const int mb = blockIdx.y * 64;
  const int nb = blockIdx.x * 128;
  const int m0 = mb + wm * 16;
  const int n0 = nb + wn * 64;

  union { v8f v; float f[8]; } acc[4];
  const v8f vzero = {0.f, 0.f, 0.f, 0.f, 0.f, 0.f, 0.f, 0.f};
#pragma unroll
  for (int t = 0; t < 4; ++t) acc[t].v = vzero;

  for (int kk = 0; kk < K; kk += 32) {
    // --- stage A (64x32 fp32 -> hi/lo bf16, row-major) ---
#pragma unroll
    for (int i = 0; i < 8; ++i) {
      int idx = tid + i * 256;
      int ar = idx >> 5, ac = idx & 31;
      int mr = mb + ar;
      float x = (mr < rowsA0)
                    ? A0[(size_t)mr * DMODEL + kk + ac]
                    : A1[(size_t)(mr - rowsA0) * DMODEL + kk + ac];
      unsigned short hi = f2bf(x);
      sAhi[ar][ac] = hi;
      sAlo[ar][ac] = f2bf(x - bf2f(hi));
    }
    // --- stage B (32x128 fp32 -> hi/lo bf16, TRANSPOSED: sB[col][k]) ---
#pragma unroll
    for (int i = 0; i < 16; ++i) {
      int idx = tid + i * 256;
      int br = idx >> 7, bc = idx & 127;
      const float* bp = &B[(size_t)(kk + br) * DMODEL + nb + bc];
      float x = *bp;
      if (kk + 32 < K) __builtin_prefetch(bp + 32 * DMODEL, 0, 1);
      unsigned short hi = f2bf(x);
      sBhi[bc][br] = hi;
      sBlo[bc][br] = f2bf(x - bf2f(hi));
    }
    __syncthreads();

    // --- fragments (all 2x b128) + 12 WMMAs ---
    Frag16 ah, al;
    load_afrag(ah, &sAhi[wm * 16 + mlane][0], g);
    load_afrag(al, &sAlo[wm * 16 + mlane][0], g);
#pragma unroll
    for (int t = 0; t < 4; ++t) {
      const int c = wn * 64 + t * 16 + mlane;
      Frag16 bh, bl;
      load_bfrag(bh, &sBhi[c][0], g);
      load_bfrag(bl, &sBlo[c][0], g);
      acc[t].v = wmma_bf16(ah.v, bh.v, acc[t].v);
      acc[t].v = wmma_bf16(ah.v, bl.v, acc[t].v);
      acc[t].v = wmma_bf16(al.v, bh.v, acc[t].v);
    }
    __syncthreads();
  }

  // --- epilogue: bias + store (C/D layout: row = r + 8*g, col = lane&15) ---
#pragma unroll
  for (int t = 0; t < 4; ++t) {
    const int col = n0 + t * 16 + mlane;
    const float bb = bias ? bias[col] : 0.f;
    if (outF || outB) {
#pragma unroll
      for (int r = 0; r < 8; ++r) {
        int row = m0 + r + 8 * g;
        float v = acc[t].f[r] + bb;
        if (outF) outF[(size_t)row * DMODEL + col] = v;
        if (outB) outB[(size_t)row * DMODEL + col] = f2bf(v);
      }
    }
    if (outBT) {  // transposed store: rows m0+8g..+7 are contiguous in s
      union { unsigned short h[8]; uint4 q; } pk;
#pragma unroll
      for (int r = 0; r < 8; ++r) pk.h[r] = f2bf(acc[t].f[r] + bb);
      *reinterpret_cast<uint4*>(&outBT[(size_t)col * S_ALL + m0 + 8 * g]) = pk.q;
    }
  }
}

// ---------------------------------------------------------------------------
// RMSNorm (per 128-dim head) + interleaved RoPE (rows < 2048) -> bf16 rows.
// ---------------------------------------------------------------------------
__global__ __launch_bounds__(128)
void normrope_kernel(const float* __restrict__ in,
                     const float* __restrict__ normw,
                     const float* __restrict__ cosv,
                     const float* __restrict__ sinv,
                     unsigned short* __restrict__ outRow,   // [S][3072]
                     float scale)
{
  const int s = blockIdx.x;
  const int h = blockIdx.y;
  const int d = threadIdx.x;

  __shared__ float red[128];
  __shared__ float nv[128];

  float x = in[(size_t)s * DMODEL + h * HD + d];
  red[d] = x * x;
  __syncthreads();
#pragma unroll
  for (int off = 64; off > 0; off >>= 1) {
    if (d < off) red[d] += red[d + off];
    __syncthreads();
  }
  float rms = rsqrtf(red[0] * (1.0f / HD) + 1e-6f);
  float v = x * rms * normw[d];
  nv[d] = v;
  __syncthreads();

  float outv = v;
  if (s < S_IMG) {
    float c  = cosv[(size_t)s * HD + d];
    float sn = sinv[(size_t)s * HD + d];
    float partner = nv[d ^ 1];
    outv = (d & 1) ? (v * c + partner * sn)    // odd:  xi*cos + xr*sin
                   : (v * c - partner * sn);   // even: xr*cos - xi*sin
  }
  outRow[(size_t)s * DMODEL + h * HD + d] = f2bf(outv * scale);
}

// ---------------------------------------------------------------------------
// Flash attention: 2 waves/block (two 16-query tiles of one head) sharing
// double-buffered K/V chunks streamed via global_load_async_to_lds_b128.
// Qb/Kb: bf16 [S][3072]; Vt: bf16 [H*128][S]. Q pre-scaled 1/sqrt(128)*log2e.
// ---------------------------------------------------------------------------
__global__ __launch_bounds__(64)
void flash_attn_kernel(const unsigned short* __restrict__ Qb,
                       const unsigned short* __restrict__ Kb,
                       const unsigned short* __restrict__ Vt,
                       float* __restrict__ O,                  // [S][3072] ws
                       float* __restrict__ outTxt)             // rows >= 2048
{
  __shared__ alignas(16) unsigned short sK[2][32][136];  // [buf][key][d]  pad
  __shared__ alignas(16) unsigned short sV[2][128][40];  // [buf][d][key] pad
  __shared__ alignas(16) unsigned short sP[2][16][32];   // per-wave probs

  const int tid  = threadIdx.x;
  const int w    = tid >> 5;          // wave 0/1
  const int lane = tid & 31;
  const int m    = lane & 15;
  const int g    = lane >> 4;
  const int h    = blockIdx.y;
  const int q0   = (blockIdx.x * 2 + w) * 16;

  const unsigned short* KbH = Kb + h * HD;
  const unsigned short* VtH = Vt + (size_t)h * HD * S_ALL;

  // Issue one 32-key chunk: wave0 -> K (32x128), wave1 -> V (128x32); 16 b128/wave.
  auto issue_chunk = [&](int kk, int b) {
    if (w == 0) {
#pragma unroll
      for (int i = 0; i < 16; ++i) {
        int flat = i * 32 + lane;            // 0..511
        int key = flat >> 4, dseg = flat & 15;
        async_copy_b128(&sK[b][key][dseg * 8],
                        KbH + (size_t)(kk + key) * DMODEL + dseg * 8);
      }
    } else {
#pragma unroll
      for (int i = 0; i < 16; ++i) {
        int flat = i * 32 + lane;            // 0..511
        int d = flat >> 2, kseg = flat & 3;
        async_copy_b128(&sV[b][d][kseg * 8],
                        VtH + (size_t)d * S_ALL + kk + kseg * 8);
      }
    }
  };

  // Q A-fragments for the 4 contraction chunks of head dim 128.
  Frag16 qa[4];
#pragma unroll
  for (int c = 0; c < 4; ++c)
    load_afrag(qa[c], &Qb[(size_t)(q0 + m) * DMODEL + h * HD + c * 32], g);

  union { v8f v; float f[8]; } acc[8];
  const v8f vzero = {0.f, 0.f, 0.f, 0.f, 0.f, 0.f, 0.f, 0.f};
#pragma unroll
  for (int t = 0; t < 8; ++t) acc[t].v = vzero;

  float mrun[8], lrun[8];
#pragma unroll
  for (int r = 0; r < 8; ++r) { mrun[r] = -1e30f; lrun[r] = 0.f; }

  issue_chunk(0, 0);

  const int NIT = S_ALL / 32;
  for (int it = 0; it < NIT; ++it) {
    const int b = it & 1;
    wait_async0();                 // chunk `it` landed in LDS
    __syncthreads();               // both waves' halves visible
    if (it + 1 < NIT) issue_chunk((it + 1) * 32, b ^ 1);  // prefetch next

    // ---- scores: two 16-key subtiles from sK[b] ----
    union { v8f v; float f[8]; } sc[2];
    sc[0].v = vzero; sc[1].v = vzero;
#pragma unroll
    for (int u = 0; u < 2; ++u) {
#pragma unroll
      for (int c = 0; c < 4; ++c) {
        Frag16 kb;
        load_bfrag(kb, &sK[b][u * 16 + m][c * 32], g);
        sc[u].v = wmma_bf16(qa[c].v, kb.v, sc[u].v);
      }
    }

    // ---- online softmax (rows r+8g, 16 lanes per row-group) ----
    float rmax[8], mnew[8], corr[8], p0[8], p1[8], psum[8];
#pragma unroll
    for (int r = 0; r < 8; ++r) rmax[r] = fmaxf(sc[0].f[r], sc[1].f[r]);
#pragma unroll
    for (int r = 0; r < 8; ++r) {
#pragma unroll
      for (int mask = 1; mask < 16; mask <<= 1)
        rmax[r] = fmaxf(rmax[r], __shfl_xor(rmax[r], mask, 32));
    }
#pragma unroll
    for (int r = 0; r < 8; ++r) {
      mnew[r] = fmaxf(mrun[r], rmax[r]);
      corr[r] = exp2f(mrun[r] - mnew[r]);
      p0[r]   = exp2f(sc[0].f[r] - mnew[r]);
      p1[r]   = exp2f(sc[1].f[r] - mnew[r]);
      psum[r] = p0[r] + p1[r];
      mrun[r] = mnew[r];
    }
#pragma unroll
    for (int r = 0; r < 8; ++r) {
#pragma unroll
      for (int mask = 1; mask < 16; mask <<= 1)
        psum[r] += __shfl_xor(psum[r], mask, 32);
      lrun[r] = lrun[r] * corr[r] + psum[r];
    }
#pragma unroll
    for (int t = 0; t < 8; ++t)
#pragma unroll
      for (int r = 0; r < 8; ++r) acc[t].f[r] *= corr[r];

    // ---- probs: C layout -> LDS -> A layout (2x ds_load_b128) ----
#pragma unroll
    for (int r = 0; r < 8; ++r) {
      sP[w][r + 8 * g][m]      = f2bf(p0[r]);
      sP[w][r + 8 * g][16 + m] = f2bf(p1[r]);
    }
    __syncthreads();               // order sP store -> load (and pace waves)
    Frag16 pa;
    load_afrag(pa, &sP[w][m][0], g);

    // ---- P * V from sV[b] ----
#pragma unroll
    for (int t = 0; t < 8; ++t) {
      Frag16 vb;
      load_bfrag(vb, &sV[b][t * 16 + m][0], g);
      acc[t].v = wmma_bf16(pa.v, vb.v, acc[t].v);
    }
  }

  // ---- epilogue: normalize and store ----
#pragma unroll
  for (int t = 0; t < 8; ++t) {
#pragma unroll
    for (int r = 0; r < 8; ++r) {
      int row = q0 + r + 8 * g;
      int col = h * HD + t * 16 + m;
      float val = acc[t].f[r] / lrun[r];
      if (row < S_IMG) O[(size_t)row * DMODEL + col] = val;
      else             outTxt[(size_t)(row - S_IMG) * DMODEL + col] = val;
    }
  }
}

// ---------------------------------------------------------------------------
extern "C" void kernel_launch(void* const* d_in, const int* in_sizes, int n_in,
                              void* d_out, int out_size, void* d_ws, size_t ws_size,
                              hipStream_t stream) {
  const float* hidden  = (const float*)d_in[0];   // [2048, 3072]
  const float* encoder = (const float*)d_in[1];   // [256, 3072]
  const float* ropeCos = (const float*)d_in[2];   // [2048, 128]
  const float* ropeSin = (const float*)d_in[3];   // [2048, 128]
  const float* wq = (const float*)d_in[4];
  const float* bq = (const float*)d_in[5];
  const float* wk = (const float*)d_in[6];
  const float* bk = (const float*)d_in[7];
  const float* wv = (const float*)d_in[8];
  const float* bv = (const float*)d_in[9];
  const float* nqw = (const float*)d_in[10];
  const float* nkw = (const float*)d_in[11];
  const float* wo = (const float*)d_in[12];
  const float* bo = (const float*)d_in[13];

  float* out = (float*)d_out;                 // img [2048*3072] then txt [256*3072]

  const size_t SD = (size_t)S_ALL * DMODEL;   // 7,077,888 elements
  float* Qf = (float*)d_ws;
  float* Kf = Qf + SD;
  float* Of = Kf + SD;
  unsigned short* Qb = (unsigned short*)(Of + SD);
  unsigned short* Kb = Qb + SD;
  unsigned short* Vt = Kb + SD;               // [H*128][S_ALL]

  const float qscale = 0.08838834764831845f * 1.4426950408889634f; // 1/sqrt(128)*log2(e)

  dim3 gBlk(256);
  dim3 gGridFull(DMODEL / 128, S_ALL / 64);   // 24 x 36
  dim3 gGridImg(DMODEL / 128, S_IMG / 64);    // 24 x 32

  // Q, K, V projections (X = concat(hidden, encoder) handled in-kernel).
  gemm_splitbf16_kernel<<<gGridFull, gBlk, 0, stream>>>(
      hidden, encoder, S_IMG, wq, bq, Qf, nullptr, nullptr, S_ALL, DMODEL, DMODEL);
  gemm_splitbf16_kernel<<<gGridFull, gBlk, 0, stream>>>(
      hidden, encoder, S_IMG, wk, bk, Kf, nullptr, nullptr, S_ALL, DMODEL, DMODEL);
  gemm_splitbf16_kernel<<<gGridFull, gBlk, 0, stream>>>(
      hidden, encoder, S_IMG, wv, bv, nullptr, nullptr, Vt, S_ALL, DMODEL, DMODEL);

  // RMSNorm + RoPE -> bf16 rows (Q pre-scaled for base-2 softmax).
  normrope_kernel<<<dim3(S_ALL, NHEADS), 128, 0, stream>>>(
      Qf, nqw, ropeCos, ropeSin, Qb, qscale);
  normrope_kernel<<<dim3(S_ALL, NHEADS), 128, 0, stream>>>(
      Kf, nkw, ropeCos, ropeSin, Kb, 1.0f);

  // Flash attention (async-pipelined K/V); txt rows written directly to d_out.
  flash_attn_kernel<<<dim3(S_ALL / 32, NHEADS), 64, 0, stream>>>(
      Qb, Kb, Vt, Of, out + (size_t)S_IMG * DMODEL);

  // Output projection for img rows.
  gemm_splitbf16_kernel<<<gGridImg, gBlk, 0, stream>>>(
      Of, Of, S_ALL, wo, bo, out, nullptr, nullptr, S_IMG, DMODEL, DMODEL);
}
